// UKNFilter_81389630259826
// MI455X (gfx1250) — matrix operand order repaired and checked by reading, not compiled, not run on hardware
//
#include <hip/hip_runtime.h>

// ---------------------------------------------------------------------------
// UKF + neural-gain GRU filter for MI455X (gfx1250, wave32, WMMA).
// One wave owns 32 batch elements for all 64 timesteps. Scalar 4x4/2x2 UKF
// algebra per lane in f32; all gain-net layers via v_wmma_f32_16x16x32_f16
// with f16 operands staged in LDS, f32 accumulation, biases folded into C.
// ---------------------------------------------------------------------------

typedef __attribute__((ext_vector_type(16))) _Float16 v16h;
typedef __attribute__((ext_vector_type(8)))  float    v8f;

#define WAVES_PER_BLOCK 4
#define TLEN 64

namespace {
constexpr float kDT = 0.05f, kCD = 0.25f, kJIT = 1e-6f, kFEPS = 1e-8f;
constexpr float kDKS = 0.1f, kRHOT = 0.3f;
constexpr float kSQRTC = 1.6f;       // sqrt(cc), cc = 4 + lam, lam = 0.8^2*4-4
constexpr float kWM0   = -0.5625f;   // lam/cc
constexpr float kWI    = 0.1953125f; // 1/(2cc)
constexpr float kWC0   = 1.7975f;    // lam/cc + (1 - 0.8^2 + 2)
constexpr float kQd[4] = {1e-5f, 1e-4f, 1e-6f, 1e-6f};
constexpr float kR     = 0.0025f;    // 0.05^2
}

struct Args {
  const float *y, *u, *z0, *P0;
  const float *ln_g, *ln_b;
  const float *enc_w1, *enc_b1, *enc_w2, *enc_b2;
  const float *gru_wih, *gru_whh, *gru_bih, *gru_bhh;
  const float *trunk_w, *trunk_b;
  const float *dk_w1, *dk_b1, *dk_w2, *dk_b2;
  const float *gate_w, *gate_b;
  float *out;
  int batch;
};

__device__ __forceinline__ float sigmoidf_(float x) { return 1.0f / (1.0f + __expf(-x)); }

// A operand (16x32 f16, MxK): lane half selects K sub-block per ISA layout.
// frag[j], j<8 : K = half*8 + j ; j>=8 : K = 16 + half*8 + (j-8); M = lane&15.
__device__ __forceinline__ v16h loadA(const _Float16 (*buf)[32], int m, int lane) {
  const int half = lane >> 4;
  const _Float16* r = &buf[m * 16 + (lane & 15)][0];
  v16h a;
#pragma unroll
  for (int j = 0; j < 8; ++j) a[j] = r[half * 8 + j];
#pragma unroll
  for (int j = 0; j < 8; ++j) a[8 + j] = r[16 + half * 8 + j];
  return a;
}

// B operand fragment, pre-swizzled in LDS: wf[lane][j] = W[K=(lane/16)*8+(j&7)+(j>>3)*16][N=lane&15].
__device__ __forceinline__ v16h loadB(const _Float16 (*wf)[16], int lane) {
  const _Float16* p = &wf[lane][0];
  v16h b;
#pragma unroll
  for (int j = 0; j < 16; ++j) b[j] = p[j];
  return b;
}

// C/D f32 layout: element r of lane L -> row = 8*(L/16)+r, col = L&15.
__device__ __forceinline__ v8f biasD(const float* bias, int colbase, int lane) {
  const float bv = bias[colbase + (lane & 15)];
  v8f c;
#pragma unroll
  for (int r = 0; r < 8; ++r) c[r] = bv;
  return c;
}

__device__ __forceinline__ void storeD(_Float16 (*buf)[32], int m, int ncol, v8f d, int lane) {
  const int col = ncol * 16 + (lane & 15);
  const int r0 = m * 16 + ((lane >> 4) << 3);
#pragma unroll
  for (int r = 0; r < 8; ++r) buf[r0 + r][col] = (_Float16)d[r];
}

__device__ __forceinline__ v8f wmma16(v16h a, v16h b, v8f c) {
  return __builtin_amdgcn_wmma_f32_16x16x32_f16(false, a, false, b, (short)0, c, false, false);
}

__device__ __forceinline__ void ffn_layer(const _Float16 (*src)[32], _Float16 (*dst)[32],
                                          const _Float16 (*wf0)[16], const _Float16 (*wf1)[16],
                                          const float* bias, bool relu, int lane) {
#pragma unroll
  for (int m = 0; m < 2; ++m) {
    v16h a = loadA(src, m, lane);
#pragma unroll
    for (int n = 0; n < 2; ++n) {
      v8f acc = biasD(bias, n * 16, lane);
      acc = wmma16(a, loadB(n ? wf1 : wf0, lane), acc);
      if (relu) {
#pragma unroll
        for (int r = 0; r < 8; ++r) acc[r] = fmaxf(acc[r], 0.0f);
      }
      storeD(dst, m, n, acc, lane);
    }
  }
}

__launch_bounds__(WAVES_PER_BLOCK * 32, 1)
__global__ void uknf_fused_kernel(Args A) {
  // Weight B-fragments, f16, pre-swizzled into WMMA layout.
  // tiles: 0-1 enc1 | 2-3 enc2 | 4-9 wih | 10-15 whh | 16-17 trunk | 18-19 dk1 | 20 dk2
  __shared__ __align__(16) _Float16 w_frag[21][32][16];
  __shared__ __align__(16) _Float16 stA[WAVES_PER_BLOCK][32][32];  // activation rows (f16)
  __shared__ __align__(16) _Float16 stH[WAVES_PER_BLOCK][32][32];  // hidden rows / dk rows
  __shared__ float s_eb1[32], s_eb2[32], s_bih[96], s_bhh[96];
  __shared__ float s_tb[32], s_db1[32], s_db2[16], s_gw[64], s_gb[2], s_lng[20], s_lnb[20];

  const int tid = threadIdx.x;
  const int wv = tid >> 5;
  const int lane = tid & 31;

  // ----- one-time weight swizzle (f32 -> f16, WMMA B layout) -----
  for (int idx = tid; idx < 21 * 512; idx += blockDim.x) {
    const int tile = idx >> 9, rem = idx & 511, ln = rem >> 4, j = rem & 15;
    const int k = ((ln >> 4) << 3) + (j & 7) + ((j >> 3) << 4);
    const float* W; int Kin, Nout, nb;
    if (tile < 2)       { W = A.enc_w1;  Kin = 20; Nout = 32; nb = tile * 16; }
    else if (tile < 4)  { W = A.enc_w2;  Kin = 32; Nout = 32; nb = (tile - 2) * 16; }
    else if (tile < 10) { W = A.gru_wih; Kin = 32; Nout = 96; nb = (tile - 4) * 16; }
    else if (tile < 16) { W = A.gru_whh; Kin = 32; Nout = 96; nb = (tile - 10) * 16; }
    else if (tile < 18) { W = A.trunk_w; Kin = 32; Nout = 32; nb = (tile - 16) * 16; }
    else if (tile < 20) { W = A.dk_w1;   Kin = 32; Nout = 32; nb = (tile - 18) * 16; }
    else                { W = A.dk_w2;   Kin = 32; Nout = 8;  nb = 0; }
    const int n = nb + (ln & 15);
    const float v = (k < Kin && n < Nout) ? W[n * Kin + k] : 0.0f;
    w_frag[tile][ln][j] = (_Float16)v;
  }
  if (tid < 32) { s_eb1[tid] = A.enc_b1[tid]; s_eb2[tid] = A.enc_b2[tid];
                  s_tb[tid] = A.trunk_b[tid]; s_db1[tid] = A.dk_b1[tid]; }
  if (tid < 96) { s_bih[tid] = A.gru_bih[tid]; s_bhh[tid] = A.gru_bhh[tid]; }
  if (tid < 16) { s_db2[tid] = (tid < 8) ? A.dk_b2[tid] : 0.0f; }
  if (tid < 64) { s_gw[tid] = A.gate_w[tid]; }
  if (tid < 2)  { s_gb[tid] = A.gate_b[tid]; }
  if (tid < 20) { s_lng[tid] = A.ln_g[tid]; s_lnb[tid] = A.ln_b[tid]; }
  __syncthreads();

  _Float16 (*bufA)[32] = stA[wv];
  _Float16 (*bufH)[32] = stH[wv];

  // ----- persistent per-lane filter state -----
  const long b = (long)blockIdx.x * (WAVES_PER_BLOCK * 32) + tid;
  float z[4], Pm[4][4];
#pragma unroll
  for (int i = 0; i < 4; ++i) z[i] = A.z0[b * 4 + i];
#pragma unroll
  for (int i = 0; i < 4; ++i)
#pragma unroll
    for (int j = 0; j < 4; ++j) Pm[i][j] = A.P0[b * 16 + i * 4 + j];

  v8f h_tile[2][2];  // GRU hidden in WMMA C/D layout
#pragma unroll
  for (int m = 0; m < 2; ++m)
#pragma unroll
    for (int n = 0; n < 2; ++n)
#pragma unroll
      for (int r = 0; r < 8; ++r) h_tile[m][n][r] = 0.0f;

#pragma unroll 1
  for (int t = 0; t < TLEN; ++t) {
    const float y0 = A.y[(b * TLEN + t) * 2 + 0];
    const float y1 = A.y[(b * TLEN + t) * 2 + 1];
    const float ut = A.u[b * TLEN + t];
    if (t + 1 < TLEN) __builtin_prefetch(&A.y[(b * TLEN + t + 1) * 2], 0, 1);

    // ---------- scalar UKF predict (per lane) ----------
    float Ps[4][4], L[4][4];
#pragma unroll
    for (int i = 0; i < 4; ++i)
#pragma unroll
      for (int j = 0; j < 4; ++j)
        Ps[i][j] = 0.5f * (Pm[i][j] + Pm[j][i]) + (i == j ? kJIT : 0.0f);
#pragma unroll
    for (int j = 0; j < 4; ++j) {
      float s = Ps[j][j];
#pragma unroll
      for (int k2 = 0; k2 < 4; ++k2) if (k2 < j) s -= L[j][k2] * L[j][k2];
      const float d = sqrtf(fmaxf(s, 1e-20f));
      L[j][j] = d;
      const float inv = 1.0f / d;
#pragma unroll
      for (int i = 0; i < 4; ++i) if (i > j) {
        float v = Ps[i][j];
#pragma unroll
        for (int k2 = 0; k2 < 4; ++k2) if (k2 < j) v -= L[i][k2] * L[j][k2];
        L[i][j] = v * inv;
      }
    }
    // sigma points + dynamics
    float Xs[9][4];
#pragma unroll
    for (int s = 0; s < 9; ++s) {
      float sg[4];
#pragma unroll
      for (int i = 0; i < 4; ++i) sg[i] = z[i];
      if (s >= 1 && s <= 4) {
        const int c = s - 1;
#pragma unroll
        for (int i = 0; i < 4; ++i) sg[i] += kSQRTC * L[i][c];
      } else if (s >= 5) {
        const int c = s - 5;
#pragma unroll
        for (int i = 0; i < 4; ++i) sg[i] -= kSQRTC * L[i][c];
      }
      const float p = sg[0], v = sg[1], kk = sg[2], al = sg[3];
      Xs[s][0] = p + kDT * v;
      Xs[s][1] = v + kDT * (-kCD * v - kk * p - al * p * p * p + ut);
      Xs[s][2] = kk;
      Xs[s][3] = al;
    }
    float zp[4];
#pragma unroll
    for (int i = 0; i < 4; ++i) {
      float acc = 0.0f;
#pragma unroll
      for (int s = 1; s < 9; ++s) acc += Xs[s][i];
      zp[i] = kWM0 * Xs[0][i] + kWI * acc;
    }
    float dX[9][4];
#pragma unroll
    for (int s = 0; s < 9; ++s)
#pragma unroll
      for (int i = 0; i < 4; ++i) dX[s][i] = Xs[s][i] - zp[i];
    float Mm[4][4];
#pragma unroll
    for (int i = 0; i < 4; ++i)
#pragma unroll
      for (int j = i; j < 4; ++j) {
        float acc = 0.0f;
#pragma unroll
        for (int s = 1; s < 9; ++s) acc += dX[s][i] * dX[s][j];
        const float v = kWC0 * dX[0][i] * dX[0][j] + kWI * acc;
        Mm[i][j] = v; Mm[j][i] = v;
      }
    float Pp[4][4];
#pragma unroll
    for (int i = 0; i < 4; ++i)
#pragma unroll
      for (int j = 0; j < 4; ++j)
        Pp[i][j] = Mm[i][j] + (i == j ? kQd[i] + kJIT : 0.0f);
    const float s00 = Mm[0][0] + kR + kJIT, s10 = Mm[1][0], s11 = Mm[1][1] + kR + kJIT;
    // chol(S + JIT)
    const float lS00 = sqrtf(fmaxf(s00 + kJIT, 1e-20f));
    const float lS10 = s10 / lS00;
    const float lS11 = sqrtf(fmaxf(s11 + kJIT - lS10 * lS10, 1e-20f));
    // K_ukf = P_zy S^{-1} via fwd/back substitution per row
    float Kk[4][2];
#pragma unroll
    for (int i = 0; i < 4; ++i) {
      const float w0 = Mm[i][0] / lS00;
      const float w1 = (Mm[i][1] - lS10 * w0) / lS11;
      const float x1 = w1 / lS11;
      const float x0 = (w0 - lS10 * x1) / lS00;
      Kk[i][0] = x0; Kk[i][1] = x1;
    }
    const float nu0 = y0 - zp[0], nu1 = y1 - zp[1];
    const float tn0 = nu0 / lS00, tn1 = (nu1 - lS10 * tn0) / lS11;
    const float nis = tn0 * tn0 + tn1 * tn1;
    const float logdetS = 2.0f * (logf(lS00 + kFEPS) + logf(lS11 + kFEPS));
    // C = L_t^{-1} P_ty (chol(S))^{-T}
    const float lt00 = sqrtf(fmaxf(Pp[2][2] + kJIT, 1e-20f));
    const float lt10 = 0.5f * (Pp[3][2] + Pp[2][3]) / lt00;
    const float lt11 = sqrtf(fmaxf(Pp[3][3] + kJIT - lt10 * lt10, 1e-20f));
    float Ct[2][2];
#pragma unroll
    for (int j = 0; j < 2; ++j) {
      Ct[0][j] = Mm[2][j] / lt00;
      Ct[1][j] = (Mm[3][j] - lt10 * Ct[0][j]) / lt11;
    }
    const float i00 = 1.0f / lS00, i10 = -lS10 / (lS00 * lS11), i11 = 1.0f / lS11;
    float Cm[2][2];
#pragma unroll
    for (int i = 0; i < 2; ++i) {
      Cm[i][0] = Ct[i][0] * i00 + Ct[i][1] * i10;
      Cm[i][1] = Ct[i][1] * i11;
    }
    float dzu[4];
#pragma unroll
    for (int i = 0; i < 4; ++i) dzu[i] = Kk[i][0] * nu0 + Kk[i][1] * nu1;

    float feat[20];
    feat[0] = tn0; feat[1] = tn1; feat[2] = nis; feat[3] = logdetS;
#pragma unroll
    for (int i = 0; i < 4; ++i) feat[4 + i] = zp[i] - z[i];
#pragma unroll
    for (int i = 0; i < 4; ++i) feat[8 + i] = logf(Pp[i][i] + kFEPS);
    feat[12] = Cm[0][0]; feat[13] = Cm[0][1]; feat[14] = Cm[1][0]; feat[15] = Cm[1][1];
#pragma unroll
    for (int i = 0; i < 4; ++i) feat[16 + i] = dzu[i];

    // LayerNorm over 20 feats, stage f16 rows (K padded to 32 with zeros)
    float mu = 0.0f;
#pragma unroll
    for (int c = 0; c < 20; ++c) mu += feat[c];
    mu *= (1.0f / 20.0f);
    float var = 0.0f;
#pragma unroll
    for (int c = 0; c < 20; ++c) { const float d = feat[c] - mu; var += d * d; }
    var *= (1.0f / 20.0f);
    const float rinv = rsqrtf(var + 1e-5f);
    {
      _Float16* frow = &bufA[lane][0];
#pragma unroll
      for (int c = 0; c < 20; ++c)
        frow[c] = (_Float16)((feat[c] - mu) * rinv * s_lng[c] + s_lnb[c]);
#pragma unroll
      for (int c = 20; c < 32; ++c) frow[c] = (_Float16)0.0f;
    }
    // stage hidden state rows (f16) from register C/D tiles
#pragma unroll
    for (int m = 0; m < 2; ++m)
#pragma unroll
      for (int n = 0; n < 2; ++n) storeD(bufH, m, n, h_tile[m][n], lane);
    __builtin_amdgcn_wave_barrier();

    // ---------- WMMA gain net ----------
    ffn_layer(bufA, bufA, w_frag[0], w_frag[1], s_eb1, true, lane);   // enc1
    __builtin_amdgcn_wave_barrier();
    ffn_layer(bufA, bufA, w_frag[2], w_frag[3], s_eb2, true, lane);   // enc2
    __builtin_amdgcn_wave_barrier();

    // GRU: per (m,n) compute 6 WMMA tiles and the gated update in D layout
#pragma unroll
    for (int m = 0; m < 2; ++m) {
      v16h ax = loadA(bufA, m, lane);
      v16h ah = loadA(bufH, m, lane);
#pragma unroll
      for (int n = 0; n < 2; ++n) {
        v8f gir = wmma16(ax, loadB(w_frag[4 + 0 + n], lane), biasD(s_bih, 0 + n * 16, lane));
        v8f giz = wmma16(ax, loadB(w_frag[4 + 2 + n], lane), biasD(s_bih, 32 + n * 16, lane));
        v8f gin = wmma16(ax, loadB(w_frag[4 + 4 + n], lane), biasD(s_bih, 64 + n * 16, lane));
        v8f ghr = wmma16(ah, loadB(w_frag[10 + 0 + n], lane), biasD(s_bhh, 0 + n * 16, lane));
        v8f ghz = wmma16(ah, loadB(w_frag[10 + 2 + n], lane), biasD(s_bhh, 32 + n * 16, lane));
        v8f ghn = wmma16(ah, loadB(w_frag[10 + 4 + n], lane), biasD(s_bhh, 64 + n * 16, lane));
#pragma unroll
        for (int e = 0; e < 8; ++e) {
          const float r  = sigmoidf_(gir[e] + ghr[e]);
          const float zz = sigmoidf_(giz[e] + ghz[e]);
          const float nn = tanhf(gin[e] + r * ghn[e]);
          h_tile[m][n][e] = (1.0f - zz) * nn + zz * h_tile[m][n][e];
        }
        storeD(bufH, m, n, h_tile[m][n], lane);  // h_new rows for trunk
      }
    }
    __builtin_amdgcn_wave_barrier();

    ffn_layer(bufH, bufA, w_frag[16], w_frag[17], s_tb, true, lane);  // trunk -> t rows in bufA
    __builtin_amdgcn_wave_barrier();

    // gate head (N=2): per-lane VALU dot over staged t row
    float g0 = s_gb[0], g1 = s_gb[1];
    {
      const _Float16* trow = &bufA[lane][0];
#pragma unroll
      for (int c = 0; c < 32; ++c) {
        const float tv = (float)trow[c];
        g0 += tv * s_gw[c];
        g1 += tv * s_gw[32 + c];
      }
    }
    const float rho_x = sigmoidf_(g0);
    const float rho_t = kRHOT * sigmoidf_(g1);
    __builtin_amdgcn_wave_barrier();

    ffn_layer(bufA, bufA, w_frag[18], w_frag[19], s_db1, true, lane); // dk1
    __builtin_amdgcn_wave_barrier();
#pragma unroll
    for (int m = 0; m < 2; ++m) {                                      // dk2 (N=8, padded)
      v16h a = loadA(bufA, m, lane);
      v8f acc = wmma16(a, loadB(w_frag[20], lane), biasD(s_db2, 0, lane));
      storeD(bufH, m, 0, acc, lane);
    }
    __builtin_amdgcn_wave_barrier();

    // ---------- scalar UKF update (per lane) ----------
    float dkraw[8];
    {
      const _Float16* drow = &bufH[lane][0];
#pragma unroll
      for (int c = 0; c < 8; ++c) dkraw[c] = (float)drow[c];
    }
    float knorm2 = 0.0f;
#pragma unroll
    for (int i = 0; i < 4; ++i) knorm2 += Kk[i][0] * Kk[i][0] + Kk[i][1] * Kk[i][1];
    const float scale = kDKS * sqrtf(knorm2);
    float Kg[4][2];
#pragma unroll
    for (int i = 0; i < 4; ++i) {
      const float gate = (i < 2) ? rho_x : rho_t;
#pragma unroll
      for (int j = 0; j < 2; ++j)
        Kg[i][j] = Kk[i][j] + gate * (scale * tanhf(dkraw[i * 2 + j]));
    }
#pragma unroll
    for (int i = 0; i < 4; ++i) z[i] = zp[i] + Kg[i][0] * nu0 + Kg[i][1] * nu1;
    {
      float4* op = (float4*)(A.out + ((long)b * TLEN + t) * 4);
      *op = make_float4(z[0], z[1], z[2], z[3]);
    }
    // P_post = P_pred - K P_yz - P_zy K^T + K S K^T (S includes R + JIT)
    float T1[4][4], KSK[4][4];
#pragma unroll
    for (int i = 0; i < 4; ++i) {
      const float ks0 = Kg[i][0] * s00 + Kg[i][1] * s10;
      const float ks1 = Kg[i][0] * s10 + Kg[i][1] * s11;
#pragma unroll
      for (int j = 0; j < 4; ++j) {
        T1[i][j]  = Kg[i][0] * Mm[j][0] + Kg[i][1] * Mm[j][1];
        KSK[i][j] = ks0 * Kg[j][0] + ks1 * Kg[j][1];
      }
    }
#pragma unroll
    for (int i = 0; i < 4; ++i)
#pragma unroll
      for (int j = 0; j < 4; ++j) {
        const float a_ = Pp[i][j] - T1[i][j] - T1[j][i] + KSK[i][j];
        const float b_ = Pp[j][i] - T1[j][i] - T1[i][j] + KSK[j][i];
        Pm[i][j] = 0.5f * (a_ + b_) + (i == j ? kJIT : 0.0f);
      }
    __builtin_amdgcn_wave_barrier();
  }
}

extern "C" void kernel_launch(void* const* d_in, const int* in_sizes, int n_in,
                              void* d_out, int out_size, void* d_ws, size_t ws_size,
                              hipStream_t stream) {
  (void)n_in; (void)out_size; (void)d_ws; (void)ws_size;
  Args a;
  a.y       = (const float*)d_in[0];
  a.u       = (const float*)d_in[1];
  a.z0      = (const float*)d_in[2];
  a.P0      = (const float*)d_in[3];
  a.ln_g    = (const float*)d_in[4];
  a.ln_b    = (const float*)d_in[5];
  a.enc_w1  = (const float*)d_in[6];
  a.enc_b1  = (const float*)d_in[7];
  a.enc_w2  = (const float*)d_in[8];
  a.enc_b2  = (const float*)d_in[9];
  a.gru_wih = (const float*)d_in[10];
  a.gru_whh = (const float*)d_in[11];
  a.gru_bih = (const float*)d_in[12];
  a.gru_bhh = (const float*)d_in[13];
  a.trunk_w = (const float*)d_in[14];
  a.trunk_b = (const float*)d_in[15];
  a.dk_w1   = (const float*)d_in[16];
  a.dk_b1   = (const float*)d_in[17];
  a.dk_w2   = (const float*)d_in[18];
  a.dk_b2   = (const float*)d_in[19];
  a.gate_w  = (const float*)d_in[20];
  a.gate_b  = (const float*)d_in[21];
  a.out     = (float*)d_out;
  a.batch   = in_sizes[1] / TLEN;  // u is (B, T)

  const int threads = WAVES_PER_BLOCK * 32;        // 128 = 4 waves
  const int blocks = a.batch / threads;            // 16384 / 128 = 128
  uknf_fused_kernel<<<blocks, threads, 0, stream>>>(a);
}
